// BiMambaBlock_27599459844413
// MI455X (gfx1250) — matrix operand order repaired and checked by reading, compiled, not run on hardware
//
#include <hip/hip_runtime.h>
#include <hip/hip_fp16.h>

// ---------------------------------------------------------------------------
// BiMamba block for MI455X (gfx1250, wave32).
// GEMMs via V_WMMA_F32_16X16X32_F16 (f32 accumulate); everything else f32.
// ---------------------------------------------------------------------------

#define Bz   2
#define Sl   2048
#define DM   512          // D_MODEL
#define DI   1024         // D_INNER
#define DSn  16           // D_STATE
#define DTR  32           // DT_RANK
#define Tt   (Bz * Sl)    // 4096 tokens
#define RSC  1.0f

typedef __attribute__((ext_vector_type(16))) _Float16 v16h;
typedef __attribute__((ext_vector_type(8)))  float    v8f;
typedef __attribute__((ext_vector_type(8)))  _Float16 v8h;   // 16B, align 16

union AFrag { v16h v; unsigned int u[8]; };

__device__ __forceinline__ unsigned int ld2h(const _Float16* p) {
  unsigned int r; __builtin_memcpy(&r, p, 4); return r;
}

// ---------------------------------------------------------------------------
// TN GEMM: C[M,N] = A[M,K] (f16, row-major) * W[N,K]^T (f16, row-major), f32 out.
// Block tile 128x64, K-step 32, 256 threads (8 waves, 4x2 wave grid).
// Each wave owns a 32x32 C tile = 4 WMMA accumulators -> 4 wmma per K-step.
// M multiple of 128, N multiple of 64, K multiple of 32 (all shapes satisfy).
// ---------------------------------------------------------------------------
__global__ __launch_bounds__(256) void gemm_tn_f16(
    const _Float16* __restrict__ A, const _Float16* __restrict__ W,
    float* __restrict__ C, int M, int N, int K)
{
  // row stride 40 halfs = 80B: keeps every 8-half column 16B aligned and
  // spreads the per-lane fragment gathers across banks.
  __shared__ _Float16 sA[128][40];
  __shared__ _Float16 sB[64][40];
  const int m0   = blockIdx.x * 128;
  const int n0   = blockIdx.y * 64;
  const int tid  = threadIdx.x;
  const int lane = tid & 31;
  const int wave = tid >> 5;
  const int wm   = (wave & 3) * 32;   // 4 wave-rows of 32
  const int wn   = (wave >> 2) * 32;  // 2 wave-cols of 32
  const int l15  = lane & 15;
  const int hi   = lane >> 4;

  // cooperative tile-load coordinates: 16B vectors, 4 vectors per 32-half row
  const int lr = tid >> 2;            // 0..63
  const int lc = (tid & 3) * 8;       // half offset 0/8/16/24

  v8f acc00 = {}, acc01 = {}, acc10 = {}, acc11 = {};

  for (int k0 = 0; k0 < K; k0 += 32) {
    // A tile: 128 rows (two passes of 64); B tile: 64 rows (one pass).
    const v8h ta0 = *(const v8h*)&A[(size_t)(m0 + lr) * K + k0 + lc];
    const v8h ta1 = *(const v8h*)&A[(size_t)(m0 + 64 + lr) * K + k0 + lc];
    const v8h tb  = *(const v8h*)&W[(size_t)(n0 + lr) * K + k0 + lc];
    *(v8h*)&sA[lr][lc]      = ta0;
    *(v8h*)&sA[64 + lr][lc] = ta1;
    *(v8h*)&sB[lr][lc]      = tb;
    if (k0 + 32 < K) {   // uniform branch: EXEC stays all-ones for WMMA
      __builtin_prefetch(&A[(size_t)(m0 + lr) * K + k0 + 32 + lc], 0, 1);
      __builtin_prefetch(&A[(size_t)(m0 + 64 + lr) * K + k0 + 32 + lc], 0, 1);
      __builtin_prefetch(&W[(size_t)(n0 + lr) * K + k0 + 32 + lc], 0, 1);
    }
    __syncthreads();

    // A fragment (16x32, ISA layout: lanes0-15 K{0-7,16-23}, lanes16-31 K{8-15,24-31})
    // B fragment (32x16): lane = N col; lanes0-15 K{0-7,16-23}, lanes16-31 K{8-15,24-31}
    AFrag a0, a1, b0, b1;
#pragma unroll
    for (int v = 0; v < 8; ++v) {
      int ka = ((v & 4) ? 16 : 0) + hi * 8 + (v & 3) * 2;
      a0.u[v] = ld2h(&sA[wm + l15][ka]);
      a1.u[v] = ld2h(&sA[wm + 16 + l15][ka]);
      int kb = ((v & 4) ? 16 : 0) + hi * 8 + (v & 3) * 2;
      b0.u[v] = ld2h(&sB[wn + l15][kb]);
      b1.u[v] = ld2h(&sB[wn + 16 + l15][kb]);
    }
    acc00 = __builtin_amdgcn_wmma_f32_16x16x32_f16(false, a0.v, false, b0.v,
                                                   (short)0, acc00, false, false);
    acc01 = __builtin_amdgcn_wmma_f32_16x16x32_f16(false, a0.v, false, b1.v,
                                                   (short)0, acc01, false, false);
    acc10 = __builtin_amdgcn_wmma_f32_16x16x32_f16(false, a1.v, false, b0.v,
                                                   (short)0, acc10, false, false);
    acc11 = __builtin_amdgcn_wmma_f32_16x16x32_f16(false, a1.v, false, b1.v,
                                                   (short)0, acc11, false, false);
    __syncthreads();
  }

  // C/D layout: VGPR r -> M = r (lanes0-15) / r+8 (lanes16-31); N = lane&15
#pragma unroll
  for (int r = 0; r < 8; ++r) {
    int gm = m0 + wm + hi * 8 + r;
    size_t o  = (size_t)gm * N + n0 + wn + l15;
    size_t o2 = o + (size_t)16 * N;
    C[o]       = acc00[r];
    C[o + 16]  = acc01[r];
    C[o2]      = acc10[r];
    C[o2 + 16] = acc11[r];
  }
}

// ---------------------------------------------------------------------------
// LayerNorm helpers (D = 512, block = 256 threads, 2 elems/thread)
// ---------------------------------------------------------------------------
__device__ __forceinline__ void ln_stats(float a, float b, float2* red,
                                         float& mean, float& rstd)
{
  int tid = threadIdx.x;
  red[tid] = make_float2(a + b, a * a + b * b);
  __syncthreads();
  for (int off = 128; off > 0; off >>= 1) {
    if (tid < off) {
      red[tid].x += red[tid + off].x;
      red[tid].y += red[tid + off].y;
    }
    __syncthreads();
  }
  mean = red[0].x * (1.f / DM);
  float var = red[0].y * (1.f / DM) - mean * mean;
  rstd = rsqrtf(var + 1e-5f);
}

// pre-LN: optional reversed read (flip branch), f16 output for GEMM A
__global__ __launch_bounds__(256) void ln_pre_kernel(
    const float* __restrict__ x, const float* __restrict__ g,
    const float* __restrict__ bt, _Float16* __restrict__ out16, int flip)
{
  __shared__ float2 red[256];
  int t  = blockIdx.x;
  int bb = t >> 11, s = t & (Sl - 1);
  int st = flip ? (bb * Sl + (Sl - 1 - s)) : t;
  const float* row = x + (size_t)st * DM;
  int i0 = threadIdx.x, i1 = threadIdx.x + 256;
  float a = row[i0], b = row[i1];
  float mean, rstd;
  ln_stats(a, b, red, mean, rstd);
  size_t o = (size_t)t * DM;
  out16[o + i0] = (_Float16)((a - mean) * rstd * g[i0] + bt[i0]);
  out16[o + i1] = (_Float16)((b - mean) * rstd * g[i1] + bt[i1]);
}

// h = x + RS*mamba_out(un-flipped), then post-LN -> f32 + f16
__global__ __launch_bounds__(256) void add_ln_kernel(
    const float* __restrict__ x, const float* __restrict__ m, int flip,
    const float* __restrict__ g, const float* __restrict__ bt,
    float* __restrict__ out32, _Float16* __restrict__ out16)
{
  __shared__ float2 red[256];
  int t  = blockIdx.x;
  int bb = t >> 11, s = t & (Sl - 1);
  int mt = flip ? (bb * Sl + (Sl - 1 - s)) : t;
  int i0 = threadIdx.x, i1 = threadIdx.x + 256;
  float a = x[(size_t)t * DM + i0] + RSC * m[(size_t)mt * DM + i0];
  float b = x[(size_t)t * DM + i1] + RSC * m[(size_t)mt * DM + i1];
  float mean, rstd;
  ln_stats(a, b, red, mean, rstd);
  size_t o = (size_t)t * DM;
  float ya = (a - mean) * rstd * g[i0] + bt[i0];
  float yb = (b - mean) * rstd * g[i1] + bt[i1];
  out32[o + i0] = ya; out16[o + i0] = (_Float16)ya;
  out32[o + i1] = yb; out16[o + i1] = (_Float16)yb;
}

// y = h2 + RS*(f2+b2); final LN; out (+)= 0.5 * LN(y)
__global__ __launch_bounds__(256) void final_kernel(
    const float* __restrict__ h2, const float* __restrict__ f2,
    const float* __restrict__ b2, const float* __restrict__ g,
    const float* __restrict__ bt, float* __restrict__ out, int accum)
{
  __shared__ float2 red[256];
  int t = blockIdx.x;
  int i0 = threadIdx.x, i1 = threadIdx.x + 256;
  float a = h2[(size_t)t * DM + i0] + RSC * (f2[(size_t)t * DM + i0] + b2[i0]);
  float b = h2[(size_t)t * DM + i1] + RSC * (f2[(size_t)t * DM + i1] + b2[i1]);
  float mean, rstd;
  ln_stats(a, b, red, mean, rstd);
  size_t o = (size_t)t * DM;
  float ra = 0.5f * ((a - mean) * rstd * g[i0] + bt[i0]);
  float rb = 0.5f * ((b - mean) * rstd * g[i1] + bt[i1]);
  if (accum) { out[o + i0] += ra; out[o + i1] += rb; }
  else       { out[o + i0]  = ra; out[o + i1]  = rb; }
}

// ---------------------------------------------------------------------------
// Elementwise / Mamba-specific kernels
// ---------------------------------------------------------------------------
__global__ __launch_bounds__(256) void cvt_f16_kernel(
    const float* __restrict__ s, _Float16* __restrict__ d, int n)
{
  int i = blockIdx.x * 256 + threadIdx.x;
  if (i < n) d[i] = (_Float16)s[i];
}

// causal depthwise conv (4 taps) + bias + SiLU on u = xz[:, :DI]
__global__ __launch_bounds__(256) void conv_silu_kernel(
    const float* __restrict__ xz, const float* __restrict__ cw,
    const float* __restrict__ cb, float* __restrict__ u32,
    _Float16* __restrict__ u16)
{
  int idx = blockIdx.x * 256 + threadIdx.x;   // Tt*DI
  int d = idx & (DI - 1);
  int t = idx >> 10;
  int s = t & (Sl - 1);
  float acc = cb[d];
#pragma unroll
  for (int k = 0; k < 4; ++k) {
    int ss = s - 3 + k;
    if (ss >= 0) acc += xz[(size_t)(t - 3 + k) * (2 * DI) + d] * cw[d * 4 + k];
  }
  float r = acc / (1.f + expf(-acc));  // silu
  u32[idx] = r;
  u16[idx] = (_Float16)r;
}

// dt slice of dbl (cols 0..31) -> f16 for the dt_proj GEMM
__global__ __launch_bounds__(256) void extract_dt_kernel(
    const float* __restrict__ dbl, _Float16* __restrict__ dt16)
{
  int i = blockIdx.x * 256 + threadIdx.x;   // Tt*DTR
  int t = i >> 5, c = i & 31;
  dt16[i] = (_Float16)dbl[(size_t)t * 64 + c];
}

// delta = softplus(gemm + dt_proj_b), in place
__global__ __launch_bounds__(256) void softplus_kernel(
    float* __restrict__ dlt, const float* __restrict__ bias)
{
  int i = blockIdx.x * 256 + threadIdx.x;   // Tt*DI
  int d = i & (DI - 1);
  float v = dlt[i] + bias[d];
  dlt[i] = (v > 20.f) ? v : log1pf(expf(v));
}

// selective scan: one lane per (b, d, n); 16-lane shfl reduction over states
__global__ __launch_bounds__(256) void scan_kernel(
    const float* __restrict__ dlt, const float* __restrict__ u,
    const float* __restrict__ dbl, const float* __restrict__ A_log,
    float* __restrict__ y)
{
  int g = blockIdx.x * 256 + threadIdx.x;   // Bz*DI*DSn = 32768
  int n = g & 15;
  int d = (g >> 4) & (DI - 1);
  int b = g >> 14;
  float Adn = -expf(A_log[d * 16 + n]);
  float h = 0.f;
  const float* dl = dlt + (size_t)b * Sl * DI + d;
  const float* uu = u   + (size_t)b * Sl * DI + d;
  const float* bc = dbl + (size_t)b * Sl * 64;
  float*       yo = y   + (size_t)b * Sl * DI + d;
  for (int s = 0; s < Sl; ++s) {
    float dt = dl[(size_t)s * DI];
    float ut = uu[(size_t)s * DI];
    float Bt = bc[s * 64 + 32 + n];
    float Ct = bc[s * 64 + 48 + n];
    float dA = expf(dt * Adn);
    h = dA * h + dt * Bt * ut;
    float yv = h * Ct;
    yv += __shfl_xor(yv, 1);
    yv += __shfl_xor(yv, 2);
    yv += __shfl_xor(yv, 4);
    yv += __shfl_xor(yv, 8);
    if (n == 0) yo[(size_t)s * DI] = yv;
  }
}

// y = (ys + u*D) * silu(z)  -> f16 for out_proj GEMM
__global__ __launch_bounds__(256) void gate_kernel(
    const float* __restrict__ ys, const float* __restrict__ u,
    const float* __restrict__ Dp, const float* __restrict__ xz,
    _Float16* __restrict__ y16)
{
  int i = blockIdx.x * 256 + threadIdx.x;   // Tt*DI
  int d = i & (DI - 1);
  int t = i >> 10;
  float yv = ys[i] + u[i] * Dp[d];
  float z = xz[(size_t)t * (2 * DI) + DI + d];
  float r = yv * (z / (1.f + expf(-z)));
  y16[i] = (_Float16)r;
}

// exact GELU epilogue of FFN layer 1 -> f16
__global__ __launch_bounds__(256) void bias_gelu_kernel(
    const float* __restrict__ f1, const float* __restrict__ b1,
    _Float16* __restrict__ g16)
{
  int i = blockIdx.x * 256 + threadIdx.x;   // Tt*4*DM
  int j = i & (4 * DM - 1);
  float v = f1[i] + b1[j];
  float gl = 0.5f * v * (1.f + erff(v * 0.70710678118654752f));
  g16[i] = (_Float16)gl;
}

// ---------------------------------------------------------------------------
// Host driver
// ---------------------------------------------------------------------------
extern "C" void kernel_launch(void* const* d_in, const int* in_sizes, int n_in,
                              void* d_out, int out_size, void* d_ws, size_t ws_size,
                              hipStream_t stream)
{
  (void)in_sizes; (void)n_in; (void)out_size; (void)ws_size;
  const float* x = (const float*)d_in[0];

  struct P {
    const float *pre_g, *pre_b, *post_g, *post_b, *in_proj, *conv_w, *conv_b,
                *x_proj, *dt_w, *dt_b, *A_log, *Dp, *out_proj, *w1, *b1, *w2, *b2;
  };
  auto getP = [&](int base) {
    P p;
    p.pre_g   = (const float*)d_in[base + 0];
    p.pre_b   = (const float*)d_in[base + 1];
    p.post_g  = (const float*)d_in[base + 2];
    p.post_b  = (const float*)d_in[base + 3];
    p.in_proj = (const float*)d_in[base + 4];
    p.conv_w  = (const float*)d_in[base + 5];
    p.conv_b  = (const float*)d_in[base + 6];
    p.x_proj  = (const float*)d_in[base + 7];
    p.dt_w    = (const float*)d_in[base + 8];
    p.dt_b    = (const float*)d_in[base + 9];
    p.A_log   = (const float*)d_in[base + 10];
    p.Dp      = (const float*)d_in[base + 11];
    p.out_proj= (const float*)d_in[base + 12];
    p.w1      = (const float*)d_in[base + 13];
    p.b1      = (const float*)d_in[base + 14];
    p.w2      = (const float*)d_in[base + 15];
    p.b2      = (const float*)d_in[base + 16];
    return p;
  };
  P pf = getP(1), pr = getP(18);

  // --- workspace bump allocator (256B aligned) ---
  char* ws = (char*)d_ws;
  size_t off = 0;
  auto alloc = [&](size_t bytes) -> char* {
    char* p = ws + off;
    off += (bytes + 255) & ~(size_t)255;
    return p;
  };

  struct W16 { _Float16 *in, *xp, *dtw, *op, *w1, *w2; };
  auto allocW = [&]() {
    W16 w;
    w.in  = (_Float16*)alloc((size_t)2 * DI * DM * 2);
    w.xp  = (_Float16*)alloc((size_t)(DTR + 2 * DSn) * DI * 2);
    w.dtw = (_Float16*)alloc((size_t)DI * DTR * 2);
    w.op  = (_Float16*)alloc((size_t)DM * DI * 2);
    w.w1  = (_Float16*)alloc((size_t)4 * DM * DM * 2);
    w.w2  = (_Float16*)alloc((size_t)DM * 4 * DM * 2);
    return w;
  };
  W16 wf = allocW(), wr = allocW();

  _Float16* ha16 = (_Float16*)alloc((size_t)Tt * DM * 2);
  _Float16* dt16 = ha16;                       // alias: ha16 dead after GEMM1
  float* xz      = (float*)alloc((size_t)Tt * 2 * DI * 4);
  float* f1      = xz;                         // alias: xz dead before FFN GEMM1
  float* u32     = (float*)alloc((size_t)Tt * DI * 4);
  _Float16* u16  = (_Float16*)alloc((size_t)Tt * DI * 2);
  _Float16* y16  = (_Float16*)alloc((size_t)Tt * DI * 2);  // contiguous after u16
  _Float16* g16  = u16;                        // alias: spans dead u16+y16
  float* dbl     = (float*)alloc((size_t)Tt * 64 * 4);
  float* dlt     = (float*)alloc((size_t)Tt * DI * 4);
  float* ysc     = (float*)alloc((size_t)Tt * DI * 4);
  float* mout    = (float*)alloc((size_t)Tt * DM * 4);
  float* f2      = mout;                       // alias: mout dead after add_ln
  float* h2_32   = (float*)alloc((size_t)Tt * DM * 4);
  _Float16* h2_16= (_Float16*)alloc((size_t)Tt * DM * 2);

  auto cvt = [&](const float* s, _Float16* d, int n) {
    cvt_f16_kernel<<<(n + 255) / 256, 256, 0, stream>>>(s, d, n);
  };
  // convert all GEMM weights to f16 once (deterministic each call)
  cvt(pf.in_proj, wf.in, 2 * DI * DM);  cvt(pf.x_proj, wf.xp, (DTR + 2 * DSn) * DI);
  cvt(pf.dt_w, wf.dtw, DI * DTR);       cvt(pf.out_proj, wf.op, DM * DI);
  cvt(pf.w1, wf.w1, 4 * DM * DM);       cvt(pf.w2, wf.w2, DM * 4 * DM);
  cvt(pr.in_proj, wr.in, 2 * DI * DM);  cvt(pr.x_proj, wr.xp, (DTR + 2 * DSn) * DI);
  cvt(pr.dt_w, wr.dtw, DI * DTR);       cvt(pr.out_proj, wr.op, DM * DI);
  cvt(pr.w1, wr.w1, 4 * DM * DM);       cvt(pr.w2, wr.w2, DM * 4 * DM);

  auto branch = [&](const P& p, const W16& w, int flip, int accum) {
    ln_pre_kernel<<<Tt, 256, 0, stream>>>(x, p.pre_g, p.pre_b, ha16, flip);
    gemm_tn_f16<<<dim3(Tt / 128, (2 * DI) / 64), 256, 0, stream>>>(
        ha16, w.in, xz, Tt, 2 * DI, DM);
    conv_silu_kernel<<<(Tt * DI) / 256, 256, 0, stream>>>(
        xz, p.conv_w, p.conv_b, u32, u16);
    gemm_tn_f16<<<dim3(Tt / 128, 1), 256, 0, stream>>>(
        u16, w.xp, dbl, Tt, 64, DI);
    extract_dt_kernel<<<(Tt * DTR) / 256, 256, 0, stream>>>(dbl, dt16);
    gemm_tn_f16<<<dim3(Tt / 128, DI / 64), 256, 0, stream>>>(
        dt16, w.dtw, dlt, Tt, DI, DTR);
    softplus_kernel<<<(Tt * DI) / 256, 256, 0, stream>>>(dlt, p.dt_b);
    scan_kernel<<<(Bz * DI * DSn) / 256, 256, 0, stream>>>(
        dlt, u32, dbl, p.A_log, ysc);
    gate_kernel<<<(Tt * DI) / 256, 256, 0, stream>>>(ysc, u32, p.Dp, xz, y16);
    gemm_tn_f16<<<dim3(Tt / 128, DM / 64), 256, 0, stream>>>(
        y16, w.op, mout, Tt, DM, DI);
    add_ln_kernel<<<Tt, 256, 0, stream>>>(
        x, mout, flip, p.post_g, p.post_b, h2_32, h2_16);
    gemm_tn_f16<<<dim3(Tt / 128, (4 * DM) / 64), 256, 0, stream>>>(
        h2_16, w.w1, f1, Tt, 4 * DM, DM);
    bias_gelu_kernel<<<(Tt * 4 * DM) / 256, 256, 0, stream>>>(f1, p.b1, g16);
    gemm_tn_f16<<<dim3(Tt / 128, DM / 64), 256, 0, stream>>>(
        g16, w.w2, f2, Tt, DM, 4 * DM);
    final_kernel<<<Tt, 256, 0, stream>>>(
        h2_32, f2, p.b2, p.post_g, p.post_b, (float*)d_out, accum);
  };

  branch(pf, wf, /*flip=*/0, /*accum=*/0);
  branch(pr, wr, /*flip=*/1, /*accum=*/1);
}
